// SupConLossPositiveOnly_19361712571095
// MI455X (gfx1250) — matrix operand order repaired and checked by reading, not compiled, not run on hardware
//
#include <hip/hip_runtime.h>
#include <hip/hip_bf16.h>
#include <math.h>

// Problem constants (from reference setup_inputs)
#define B_ROWS 4096   // batch rows (features)
#define N_CLS  1000   // prototypes / classes
#define DIM    2048   // feature dim
#define INV_T  10.0f  // 1 / TEMPERATURE

#define KCHUNK 128    // K staged per async step (4 x 32-wide WMMA sub-chunks)
#define NCHUNKS (DIM / KCHUNK)  // 16

typedef __attribute__((ext_vector_type(16))) __bf16 v16bf;
typedef __attribute__((ext_vector_type(8)))  __bf16 v8bf;
typedef __attribute__((ext_vector_type(4)))  __bf16 v4bf;
typedef __attribute__((ext_vector_type(8)))  float  v8f;
typedef int v4i __attribute__((vector_size(16)));   // b128 payload type

#define AS_GLOBAL __attribute__((address_space(1)))
#define AS_LOCAL  __attribute__((address_space(3)))

#if defined(__has_builtin)
#  if __has_builtin(__builtin_amdgcn_global_load_async_to_lds_b128)
#    define HAVE_ASYNC_LDS 1
#  endif
#  if __has_builtin(__builtin_amdgcn_s_wait_asynccnt)
#    define HAVE_WAIT_ASYNC_BUILTIN 1
#  endif
#endif

#if defined(HAVE_ASYNC_LDS)
#  if defined(HAVE_WAIT_ASYNC_BUILTIN)
#    define WAIT_ASYNC(N) __builtin_amdgcn_s_wait_asynccnt(N)
#  else
#    define WAIT_ASYNC(N) asm volatile("s_wait_asynccnt %0" ::"i"(N) : "memory")
#  endif
#else
#  define WAIT_ASYNC(N)
#endif

// ---------------------------------------------------------------------------
// Pass 0 (fused): single pass over f32 inputs that
//   (a) splits each value into bf16 hi + bf16 residual lo planes, and
//   (b) produces the exact f32 inverse L2 norm per row.
// ---------------------------------------------------------------------------
__global__ __launch_bounds__(256)
void split_norm_kernel(const float* __restrict__ feat,
                       const float* __restrict__ prot,
                       __bf16* __restrict__ featHi, __bf16* __restrict__ featLo,
                       __bf16* __restrict__ protHi, __bf16* __restrict__ protLo,
                       float* __restrict__ invF, float* __restrict__ invP) {
  const int wave = (int)((blockIdx.x * blockDim.x + threadIdx.x) >> 5);
  const int lane = (int)(threadIdx.x & 31);

  const float* src;
  __bf16 *hiD, *loD;
  float* dst;
  if (wave < B_ROWS) {
    src = feat + (size_t)wave * DIM;
    hiD = featHi + (size_t)wave * DIM;
    loD = featLo + (size_t)wave * DIM;
    dst = invF + wave;
  } else if (wave < B_ROWS + N_CLS) {
    const int r = wave - B_ROWS;
    src = prot + (size_t)r * DIM;
    hiD = protHi + (size_t)r * DIM;
    loD = protLo + (size_t)r * DIM;
    dst = invP + r;
  } else {
    return;  // wave-uniform exit
  }

  float s = 0.0f;
#pragma unroll
  for (int k = lane * 4; k < DIM; k += 32 * 4) {
    float4 v = *(const float4*)(src + k);
    v4bf h, l;
    {
      __bf16 hb;
      hb = (__bf16)v.x; h[0] = hb; l[0] = (__bf16)(v.x - (float)hb);
      hb = (__bf16)v.y; h[1] = hb; l[1] = (__bf16)(v.y - (float)hb);
      hb = (__bf16)v.z; h[2] = hb; l[2] = (__bf16)(v.z - (float)hb);
      hb = (__bf16)v.w; h[3] = hb; l[3] = (__bf16)(v.w - (float)hb);
    }
    *(v4bf*)(hiD + k) = h;
    *(v4bf*)(loD + k) = l;
    s += v.x * v.x + v.y * v.y + v.z * v.z + v.w * v.w;
  }
#pragma unroll
  for (int off = 16; off > 0; off >>= 1) s += __shfl_xor(s, off, 32);

  if (lane == 0) *dst = 1.0f / fmaxf(sqrtf(s), 1e-12f);
}

// ---------------------------------------------------------------------------
// Fragment helpers.
// A fragment (16x32, MxK): lane needs one contiguous 16-elem run.
// B fragment (32x16, KxN): lane needs one contiguous 16-elem run.
// ---------------------------------------------------------------------------
__device__ __forceinline__ v16bf ldA(const __bf16* p) {
  v8bf r0 = *(const v8bf*)(p);
  v8bf r1 = *(const v8bf*)(p + 8);
  return __builtin_shufflevector(r0, r1, 0, 1, 2, 3, 4, 5, 6, 7,
                                 8, 9, 10, 11, 12, 13, 14, 15);
}
__device__ __forceinline__ v16bf ldB(const __bf16* p) {
  v8bf r0 = *(const v8bf*)(p);
  v8bf r1 = *(const v8bf*)(p + 8);
  return __builtin_shufflevector(r0, r1, 0, 1, 2, 3, 4, 5, 6, 7,
                                 8, 9, 10, 11, 12, 13, 14, 15);
}
__device__ __forceinline__ v8f wmma_bf16(v16bf a, v16bf b, v8f c) {
  return __builtin_amdgcn_wmma_f32_16x16x32_bf16(false, a, false, b,
                                                 (short)0, c, false, false);
}

// Stage one KCHUNK of the B (prototype) hi/lo planes into LDS.
// 256 threads move 32 KB: thread t owns (row = t>>2, 16B segment = t&3) for
// each of the 4 K-sub-chunks x 2 planes. Async path uses
// GLOBAL_LOAD_ASYNC_TO_LDS_B128 (ASYNCcnt), 8 issues per thread.
__device__ __forceinline__ void stage_B_chunk(
    const __bf16* __restrict__ protHi,
    const __bf16* __restrict__ protLo,
    __bf16 (*lds)[2][64][32],  // [kk][plane][row][elem]
    int nBase, int k0, int tid) {
  const int row = tid >> 2;  // 0..63 (tile-local prototype row)
  const int seg = tid & 3;   // 16-byte segment within the 32-elem row
  int rg = nBase + row;
  if (rg >= N_CLS) rg = N_CLS - 1;  // clamped; masked out in epilogue
  const size_t goff = (size_t)rg * DIM + seg * 8;
#pragma unroll
  for (int kk = 0; kk < 4; ++kk) {
    const __bf16* gh = protHi + goff + (k0 + kk * 32);
    const __bf16* gl = protLo + goff + (k0 + kk * 32);
    __bf16* lh = &lds[kk][0][row][seg * 8];
    __bf16* ll = &lds[kk][1][row][seg * 8];
#if defined(HAVE_ASYNC_LDS)
    __builtin_amdgcn_global_load_async_to_lds_b128(
        (AS_GLOBAL v4i*)(unsigned long long)gh,
        (AS_LOCAL v4i*)(unsigned long long)lh, 0, 0);
    __builtin_amdgcn_global_load_async_to_lds_b128(
        (AS_GLOBAL v4i*)(unsigned long long)gl,
        (AS_LOCAL v4i*)(unsigned long long)ll, 0, 0);
#else
    *(v8bf*)lh = *(const v8bf*)gh;   // sync fallback: reg-staged copy
    *(v8bf*)ll = *(const v8bf*)gl;
#endif
  }
}

// Per-tile epilogue: scale, stable softplus, label mask, half-wave row
// reduction, atomic accumulation into per-row sum/count.
__device__ __forceinline__ void epilogue(v8f c, int nTileBase, int mBase,
                                         int halfSel, int l16, float ip,
                                         const int* __restrict__ labels,
                                         const float* __restrict__ invF,
                                         float* __restrict__ rowSum,
                                         float* __restrict__ rowCnt) {
  const int  n      = nTileBase + l16;
  const bool nValid = (n < N_CLS);
#pragma unroll
  for (int j = 0; j < 8; ++j) {
    const int m = mBase + j + halfSel * 8;
    float sim = c[j] * invF[m] * ip * INV_T;
    int lab = nValid ? labels[(size_t)m * N_CLS + n] : 0;
    // stable softplus(-sim) = max(-sim,0) + log1p(exp(-|sim|))
    float e  = (lab > 0) ? (fmaxf(-sim, 0.0f) + log1pf(expf(-fabsf(sim)))) : 0.0f;
    float cc = (lab > 0) ? 1.0f : 0.0f;
#pragma unroll
    for (int off = 1; off < 16; off <<= 1) {
      e  += __shfl_xor(e,  off, 32);
      cc += __shfl_xor(cc, off, 32);
    }
    if (l16 == 0) {
      atomicAdd(&rowSum[m], e);
      atomicAdd(&rowCnt[m], cc);
    }
  }
}

// ---------------------------------------------------------------------------
// Pass 1: WMMA GEMM on pre-split bf16 planes (bf16x3, f32 accumulate) with
// the B operand double-buffered in LDS via async global->LDS copies.
// Each wave owns a 16x64 output tile; block = 8 waves -> 128x64 tile, so the
// LDS-staged B chunk is reused by all 8 waves (8x global-traffic reduction).
// ---------------------------------------------------------------------------
__global__ __launch_bounds__(256)
void gemm_loss_kernel(const __bf16* __restrict__ featHi,
                      const __bf16* __restrict__ featLo,
                      const __bf16* __restrict__ protHi,
                      const __bf16* __restrict__ protLo,
                      const int*   __restrict__ labels,
                      const float* __restrict__ invF,
                      const float* __restrict__ invP,
                      float* __restrict__ rowSum,
                      float* __restrict__ rowCnt) {
  __shared__ __align__(16) __bf16 ldsB[2][4][2][64][32];  // 64 KB

  const int tid     = (int)threadIdx.x;
  const int lane    = tid & 31;
  const int wav     = tid >> 5;        // 0..7
  const int halfSel = lane >> 4;       // 0: lanes 0-15, 1: lanes 16-31
  const int l16     = lane & 15;

  const int mBase = (int)blockIdx.x * 128 + wav * 16;  // grid.x = 32
  const int nBase = (int)blockIdx.y * 64;              // grid.y = 16 (N pad 1024)

  const __bf16* aHi = featHi + (size_t)(mBase + l16) * DIM + halfSel * 16;
  const __bf16* aLo = featLo + (size_t)(mBase + l16) * DIM + halfSel * 16;

  // Clamped prototype rows (for invP lookup in the epilogue).
  int r0 = nBase +  0 + l16; if (r0 >= N_CLS) r0 = N_CLS - 1;
  int r1 = nBase + 16 + l16; if (r1 >= N_CLS) r1 = N_CLS - 1;
  int r2 = nBase + 32 + l16; if (r2 >= N_CLS) r2 = N_CLS - 1;
  int r3 = nBase + 48 + l16; if (r3 >= N_CLS) r3 = N_CLS - 1;

  v8f c0 = {}, c1 = {}, c2 = {}, c3 = {};

  // Prologue: stage chunk 0 into buffer 0.
  stage_B_chunk(protHi, protLo, ldsB[0], nBase, 0, tid);

  for (int oc = 0; oc < NCHUNKS; ++oc) {
    const int  cur  = oc & 1;
    const bool more = (oc + 1) < NCHUNKS;

    // Stage next chunk into the other buffer (its readers finished at the
    // end-of-iteration barrier of oc-1).
    if (more) {
      stage_B_chunk(protHi, protLo, ldsB[cur ^ 1], nBase, (oc + 1) * KCHUNK, tid);
      WAIT_ASYNC(8);   // async loads complete in order: <=8 outstanding
    } else {           //  => this iteration's chunk is fully in LDS
      WAIT_ASYNC(0);
    }
    __syncthreads();   // all waves' staging of 'cur' visible block-wide

#pragma unroll
    for (int kk = 0; kk < 4; ++kk) {
      const int k0 = oc * KCHUNK + kk * 32;
      __builtin_prefetch(aHi + k0 + KCHUNK, 0, 3);
      __builtin_prefetch(aLo + k0 + KCHUNK, 0, 3);

      const v16bf ah = ldA(aHi + k0);
      const v16bf al = ldA(aLo + k0);

      // bf16x3: hi*hi + hi*lo + lo*hi (drop lo*lo) ~ f32 precision.
      {
        const v16bf bh = ldB(&ldsB[cur][kk][0][ 0 + l16][halfSel * 16]);
        const v16bf bl = ldB(&ldsB[cur][kk][1][ 0 + l16][halfSel * 16]);
        c0 = wmma_bf16(ah, bh, c0);
        c0 = wmma_bf16(ah, bl, c0);
        c0 = wmma_bf16(al, bh, c0);
      }
      {
        const v16bf bh = ldB(&ldsB[cur][kk][0][16 + l16][halfSel * 16]);
        const v16bf bl = ldB(&ldsB[cur][kk][1][16 + l16][halfSel * 16]);
        c1 = wmma_bf16(ah, bh, c1);
        c1 = wmma_bf16(ah, bl, c1);
        c1 = wmma_bf16(al, bh, c1);
      }
      {
        const v16bf bh = ldB(&ldsB[cur][kk][0][32 + l16][halfSel * 16]);
        const v16bf bl = ldB(&ldsB[cur][kk][1][32 + l16][halfSel * 16]);
        c2 = wmma_bf16(ah, bh, c2);
        c2 = wmma_bf16(ah, bl, c2);
        c2 = wmma_bf16(al, bh, c2);
      }
      {
        const v16bf bh = ldB(&ldsB[cur][kk][0][48 + l16][halfSel * 16]);
        const v16bf bl = ldB(&ldsB[cur][kk][1][48 + l16][halfSel * 16]);
        c3 = wmma_bf16(ah, bh, c3);
        c3 = wmma_bf16(ah, bl, c3);
        c3 = wmma_bf16(al, bh, c3);
      }
    }
    __syncthreads();   // done reading 'cur'; safe to overwrite next round
  }

  epilogue(c0, nBase +  0, mBase, halfSel, l16, invP[r0], labels, invF, rowSum, rowCnt);
  epilogue(c1, nBase + 16, mBase, halfSel, l16, invP[r1], labels, invF, rowSum, rowCnt);
  epilogue(c2, nBase + 32, mBase, halfSel, l16, invP[r2], labels, invF, rowSum, rowCnt);
  epilogue(c3, nBase + 48, mBase, halfSel, l16, invP[r3], labels, invF, rowSum, rowCnt);
}

// ---------------------------------------------------------------------------
// Pass 2: final scalar reduction (valid-row mean of per-row means).
// ---------------------------------------------------------------------------
__global__ __launch_bounds__(256)
void finalize_kernel(const float* __restrict__ rowSum,
                     const float* __restrict__ rowCnt,
                     float* __restrict__ out) {
  __shared__ float sTot[256];
  __shared__ float sCnt[256];
  const int tid = (int)threadIdx.x;

  float tot = 0.0f, nv = 0.0f;
  for (int r = tid; r < B_ROWS; r += 256) {
    float cnt = rowCnt[r];
    if (cnt > 0.0f) {
      tot += rowSum[r] / cnt;
      nv  += 1.0f;
    }
  }
  sTot[tid] = tot;
  sCnt[tid] = nv;
  __syncthreads();
#pragma unroll
  for (int s = 128; s > 0; s >>= 1) {
    if (tid < s) {
      sTot[tid] += sTot[tid + s];
      sCnt[tid] += sCnt[tid + s];
    }
    __syncthreads();
  }
  if (tid == 0) out[0] = (sCnt[0] > 0.0f) ? (sTot[0] / sCnt[0]) : 0.0f;
}

// ---------------------------------------------------------------------------
// Workspace layout (bytes):
//   [ 0,16M)  featHi   4096*2048 bf16
//   [16M,32M) featLo
//   [32M,36M) protHi   1000*2048 bf16 (4 MiB slot)
//   [36M,40M) protLo
//   [40M,..)  invF[4096], invP[1024], rowSum[4096], rowCnt[4096]  (f32)
// Requires ws_size >= ~41 MiB (working set stays L2-resident: 192 MB L2).
// ---------------------------------------------------------------------------
extern "C" void kernel_launch(void* const* d_in, const int* in_sizes, int n_in,
                              void* d_out, int out_size, void* d_ws, size_t ws_size,
                              hipStream_t stream) {
  const float* feat   = (const float*)d_in[0];  // (4096, 2048) f32
  const float* prot   = (const float*)d_in[1];  // (1000, 2048) f32
  const int*   labels = (const int*)d_in[2];    // (4096, 1000) i32

  char* base = (char*)d_ws;
  __bf16* featHi = (__bf16*)(base);
  __bf16* featLo = (__bf16*)(base + ((size_t)16 << 20));
  __bf16* protHi = (__bf16*)(base + ((size_t)32 << 20));
  __bf16* protLo = (__bf16*)(base + ((size_t)36 << 20));
  float*  invF   = (float*)(base + ((size_t)40 << 20));
  float*  invP   = invF + B_ROWS;          // 1024 slot
  float*  rowSum = invP + 1024;
  float*  rowCnt = rowSum + B_ROWS;

  // Zero the accumulators (capture-safe async memset).
  (void)hipMemsetAsync(rowSum, 0, 2 * B_ROWS * sizeof(float), stream);

  // Pass 0: fused split + norms. 5096 waves, 8 waves per block.
  const int totalRows = B_ROWS + N_CLS;
  split_norm_kernel<<<dim3((totalRows + 7) / 8), 256, 0, stream>>>(
      feat, prot, featHi, featLo, protHi, protLo, invF, invP);

  // Pass 1: WMMA GEMM + async-LDS B staging + fused loss epilogue.
  dim3 grid(B_ROWS / 128, (N_CLS + 63) / 64);  // 32 x 16
  gemm_loss_kernel<<<grid, 256, 0, stream>>>(featHi, featLo, protHi, protLo,
                                             labels, invF, invP, rowSum, rowCnt);

  // Pass 2: scalar.
  finalize_kernel<<<1, 256, 0, stream>>>(rowSum, rowCnt, (float*)d_out);
}
